// MemTransformerLM_8710193676756
// MI455X (gfx1250) — matrix operand order repaired and checked by reading, not compiled, not run on hardware
//
#include <hip/hip_runtime.h>
#include <hip/hip_bf16.h>
#include <math.h>

// ---------------------------------------------------------------------------
// Types for CDNA5 WMMA (gfx1250, wave32)
// ---------------------------------------------------------------------------
typedef __attribute__((ext_vector_type(16))) __bf16 v16bf;
typedef __attribute__((ext_vector_type(8)))  __bf16 v8bf;
typedef __attribute__((ext_vector_type(2)))  __bf16 v2bf;
typedef __attribute__((ext_vector_type(8)))  float  v8f;
typedef __attribute__((ext_vector_type(4)))  int    v4i;

#define QLEN 512
#define MLEN 512
#define KLEN 1024
#define BSZ  4
#define DMODEL 1024
#define DINNER 4096
#define NHEAD 16
#define DHEAD 64
#define NLAYER 6
#define LN_EPS 1e-5f

// ---- CDNA5 async load-to-LDS (ASYNCcnt) --------------------------------
#if __has_builtin(__builtin_amdgcn_global_load_async_to_lds_b128)
#define HAVE_ASYNC 1
#else
#define HAVE_ASYNC 0
#endif

#define GLBPTR(p) ((__attribute__((address_space(1))) v4i*)(p))
#define LDSPTR(p) ((__attribute__((address_space(3))) v4i*)(p))

__device__ __forceinline__ void async_b128(const void* g, void* l) {
#if HAVE_ASYNC
    __builtin_amdgcn_global_load_async_to_lds_b128(GLBPTR(g), LDSPTR(l), 0, 0);
#else
    (void)g; (void)l;
#endif
}
__device__ __forceinline__ void wait_async0() {
#if HAVE_ASYNC
#if __has_builtin(__builtin_amdgcn_s_wait_asynccnt)
    __builtin_amdgcn_s_wait_asynccnt(0);
#else
    asm volatile("s_wait_asynccnt 0x0" ::: "memory");
#endif
#endif
}

// ---- helpers ------------------------------------------------------------
__device__ __forceinline__ v16bf cat8(v8bf lo, v8bf hi) {
    return __builtin_shufflevector(lo, hi, 0,1,2,3,4,5,6,7,8,9,10,11,12,13,14,15);
}
__device__ __forceinline__ v8bf  ld8 (const __bf16* p) { return *(const v8bf *)p; }
__device__ __forceinline__ v16bf ld16(const __bf16* p) { return *(const v16bf*)p; }
__device__ __forceinline__ __bf16 tobf(float x) { return (__bf16)x; }
__device__ __forceinline__ unsigned pk2(float x, float y) {
    v2bf v = { (__bf16)x, (__bf16)y };
    return __builtin_bit_cast(unsigned, v);
}
__device__ __forceinline__ uint2 pk4(float4 f) {
    return make_uint2(pk2(f.x, f.y), pk2(f.z, f.w));
}
__device__ __forceinline__ v8f wmma_bf16(v16bf a, v16bf b, v8f c) {
    return __builtin_amdgcn_wmma_f32_16x16x32_bf16(false, a, false, b, (short)0, c, false, false);
}

// ---------------------------------------------------------------------------
// Embedding + sinusoidal positional encoding
// ---------------------------------------------------------------------------
__global__ __launch_bounds__(256) void embed_kernel(const int* __restrict__ data,
                                                    const float* __restrict__ emb,
                                                    float* __restrict__ h) {
    int i = blockIdx.x, b = blockIdx.y;
    int tok = data[i * BSZ + b];
    float p = (float)(QLEN - 1 - i);
    const float* e = emb + (size_t)tok * DMODEL;
    float* out = h + ((size_t)i * BSZ + b) * DMODEL;
    const float ln1e4 = 9.210340371976184f;  // ln(10000)
    for (int d = threadIdx.x; d < DMODEL; d += 256) {
        int j = (d < 512) ? d : (d - 512);
        float invf = __expf(-(2.0f * (float)j / (float)DMODEL) * ln1e4);
        float ang = p * invf;
        float pe = (d < 512) ? __sinf(ang) : __cosf(ang);
        out[d] = e[d] * 32.0f + pe;   // sqrt(1024) = 32
    }
}

// ---------------------------------------------------------------------------
// GEMM  C[M,N] = A[M,K] @ B[K,N]  (+bias, +relu), fp32 in/out, bf16 WMMA.
// Block tile 128x64, 8 waves; wave w computes rows w*16..w*16+15 x all 64 cols
// (1 A-frag reused by 4 B-frags -> 4 WMMA per wave per K-step).
// Raw fp32 tiles double-buffered in LDS via async loads (ASYNCcnt).
// ---------------------------------------------------------------------------
template<bool BIAS, bool RELU>
__global__ __launch_bounds__(256) void gemm_bf16(const float* __restrict__ A,
                                                 const float* __restrict__ B,
                                                 const float* __restrict__ bias,
                                                 float* __restrict__ C,
                                                 int M, int N, int K) {
#if HAVE_ASYNC
    __shared__ float rawA[2][128][32];   // 16KB x2
    __shared__ float rawB[2][32][64];    //  8KB x2
#endif
    __shared__ __bf16 sa[128][32];       // [m][k]  8KB
    __shared__ __bf16 sb[64][32];        // [n][k]  4KB (transposed)

    const int tid  = threadIdx.x;
    const int bm   = blockIdx.y * 128;
    const int bn   = blockIdx.x * 64;
    const int wave = tid >> 5, lane = tid & 31;
    const int lh   = lane >> 4, lo = lane & 15;

    v8f c0 = {}, c1 = {}, c2 = {}, c3 = {};

#if HAVE_ASYNC
    // prefetch K-step 0 into buffer 0
    {
        #pragma unroll
        for (int i = 0; i < 4; ++i) {           // A tile: 1024 float4
            int idx = tid + i * 256;
            int r = idx >> 3, cc = (idx & 7) * 4;
            async_b128(A + (size_t)(bm + r) * K + cc, &rawA[0][r][cc]);
        }
        #pragma unroll
        for (int i = 0; i < 2; ++i) {           // B tile: 512 float4
            int idx = tid + i * 256;
            int kr = idx >> 4, cc = (idx & 15) * 4;
            async_b128(B + (size_t)kr * N + bn + cc, &rawB[0][kr][cc]);
        }
    }
#endif

    int buf = 0;
    for (int k0 = 0; k0 < K; k0 += 32) {
        wait_async0();
        __syncthreads();   // raw tile visible to all; frag loads of prev step done

        // convert fp32 -> bf16 tiles (packed stores)
        #pragma unroll
        for (int i = 0; i < 4; ++i) {           // A: 4096 elems
            int idx = tid + i * 256;
            int r = idx >> 3, cc = (idx & 7) * 4;
#if HAVE_ASYNC
            float4 f = *(const float4*)&rawA[buf][r][cc];
#else
            float4 f = *(const float4*)(A + (size_t)(bm + r) * K + k0 + cc);
#endif
            *(uint2*)&sa[r][cc] = pk4(f);
        }
        {                                        // B: transpose pairs
            int kp = (tid >> 4) * 2, cc = (tid & 15) * 4;
#if HAVE_ASYNC
            float4 f0 = *(const float4*)&rawB[buf][kp][cc];
            float4 f1 = *(const float4*)&rawB[buf][kp + 1][cc];
#else
            float4 f0 = *(const float4*)(B + (size_t)(k0 + kp) * N + bn + cc);
            float4 f1 = *(const float4*)(B + (size_t)(k0 + kp + 1) * N + bn + cc);
#endif
            *(unsigned*)&sb[cc + 0][kp] = pk2(f0.x, f1.x);
            *(unsigned*)&sb[cc + 1][kp] = pk2(f0.y, f1.y);
            *(unsigned*)&sb[cc + 2][kp] = pk2(f0.z, f1.z);
            *(unsigned*)&sb[cc + 3][kp] = pk2(f0.w, f1.w);
        }

#if HAVE_ASYNC
        if (k0 + 32 < K) {                       // prefetch next K-step
            int kn = k0 + 32, nb = buf ^ 1;
            #pragma unroll
            for (int i = 0; i < 4; ++i) {
                int idx = tid + i * 256;
                int r = idx >> 3, cc = (idx & 7) * 4;
                async_b128(A + (size_t)(bm + r) * K + kn + cc, &rawA[nb][r][cc]);
            }
            #pragma unroll
            for (int i = 0; i < 2; ++i) {
                int idx = tid + i * 256;
                int kr = idx >> 4, cc = (idx & 15) * 4;
                async_b128(B + (size_t)(kn + kr) * N + bn + cc, &rawB[nb][kr][cc]);
            }
        }
#endif
        __syncthreads();   // bf16 tiles ready

        // fragments + 4 WMMA
        int m = wave * 16 + lo;
        int koff = lh * 8, kb = lh * 16;
        v16bf a  = ld16(&sa[m][kb]);
        v16bf b0 = ld16(&sb[lo][kb]);
        v16bf b1 = ld16(&sb[16 + lo][kb]);
        v16bf b2 = ld16(&sb[32 + lo][kb]);
        v16bf b3 = ld16(&sb[48 + lo][kb]);
        c0 = wmma_bf16(a, b0, c0);
        c1 = wmma_bf16(a, b1, c1);
        c2 = wmma_bf16(a, b2, c2);
        c3 = wmma_bf16(a, b3, c3);
        __syncthreads();   // before next convert overwrites sa/sb
        buf ^= 1;
    }

    // epilogue: D row = v + 8*(lane/16), col = lane%16 (+16*t)
    #pragma unroll
    for (int v = 0; v < 8; ++v) {
        int m = bm + wave * 16 + v + 8 * lh;
        float* crow = C + (size_t)m * N + bn;
        float x0 = c0[v], x1 = c1[v], x2 = c2[v], x3 = c3[v];
        if (BIAS) {
            x0 += bias[bn + lo];      x1 += bias[bn + 16 + lo];
            x2 += bias[bn + 32 + lo]; x3 += bias[bn + 48 + lo];
        }
        if (RELU) {
            x0 = fmaxf(x0, 0.0f); x1 = fmaxf(x1, 0.0f);
            x2 = fmaxf(x2, 0.0f); x3 = fmaxf(x3, 0.0f);
        }
        crow[lo]      = x0;
        crow[16 + lo] = x1;
        crow[32 + lo] = x2;
        crow[48 + lo] = x3;
    }
}

// ---------------------------------------------------------------------------
// Fused flash attention. Block = 4 waves (128 thr); each wave owns one
// 16-query tile; all waves share the staged K/V chunk (amortizes KV traffic).
// Online softmax per 16-lane C-fragment half; causal mask j <= i + MLEN.
// KV chunks double-buffered in LDS via async loads.
// ---------------------------------------------------------------------------
__global__ __launch_bounds__(128) void attn_kernel(const float* __restrict__ qbuf,
                                                   const float* __restrict__ kvbuf,
                                                   float* __restrict__ obuf) {
#if HAVE_ASYNC
    __shared__ float kvs[2][32][128];   // raw fp32 kv chunk, 16KB x2
#endif
    __shared__ __bf16 ql[4][16][64];    // per-wave q tile
    __shared__ __bf16 kl[32][64];       // K chunk [j][d] bf16
    __shared__ __bf16 vt[64][32];       // V chunk^T [d][j] bf16
    __shared__ __bf16 pl[4][16][32];    // per-wave probs

    const int i0b  = blockIdx.x * 64;
    const int b    = blockIdx.y;
    const int hn   = blockIdx.z;
    const int tid  = threadIdx.x;
    const int wave = tid >> 5, lane = tid & 31;
    const int lh = lane >> 4, lo = lane & 15;
    const int koff = lh * 8, kb = lh * 16;
    const int i0 = i0b + wave * 16;

    // stage Q tile (per wave): lane covers half a row
    {
        int r = lane >> 1, part = (lane & 1) * 32;
        const float* src = qbuf + ((size_t)(i0 + r) * BSZ + b) * (NHEAD * DHEAD)
                         + hn * DHEAD + part;
        #pragma unroll
        for (int d = 0; d < 32; d += 4)
            *(uint2*)&ql[wave][r][part + d] = pk4(*(const float4*)(src + d));
    }
    __syncthreads();

    v16bf aq0 = ld16(&ql[wave][lo][kb]);
    v16bf aq1 = ld16(&ql[wave][lo][32 + kb]);

    v8f o0 = {}, o1 = {}, o2 = {}, o3 = {};
    float rmax[8], rsum[8];
    #pragma unroll
    for (int v = 0; v < 8; ++v) { rmax[v] = -3.0e38f; rsum[v] = 0.0f; }
    const float scale = 0.125f;

    int jcmax = (i0b + 63 + MLEN) >> 5;            // block-uniform bound
    if (jcmax > (KLEN / 32) - 1) jcmax = (KLEN / 32) - 1;

    const size_t kvrow = (size_t)BSZ * NHEAD * 2 * DHEAD;  // floats per j step
    const float* kvbase = kvbuf + (size_t)b * (NHEAD * 2 * DHEAD) + hn * (2 * DHEAD);

#if HAVE_ASYNC
    {   // prefetch chunk 0: 1024 float4 over 128 threads
        #pragma unroll
        for (int i = 0; i < 8; ++i) {
            int idx = tid + i * 128;
            int key = idx >> 5, d4 = (idx & 31) * 4;
            async_b128(kvbase + (size_t)key * kvrow + d4, &kvs[0][key][d4]);
        }
    }
#endif

    int buf = 0;
    for (int jc = 0; jc <= jcmax; ++jc) {
        int j0 = jc * 32;
        wait_async0();
        __syncthreads();

        // convert chunk -> kl (rows) and vt (transposed), packed stores
        {
            int jp = (tid & 15) * 2;               // key pair
            int ds = (tid >> 4) * 8;               // 8-dim section
            #pragma unroll
            for (int dd = 0; dd < 8; dd += 4) {
                int d = ds + dd;
#if HAVE_ASYNC
                float4 k0 = *(const float4*)&kvs[buf][jp][d];
                float4 k1 = *(const float4*)&kvs[buf][jp + 1][d];
                float4 v0 = *(const float4*)&kvs[buf][jp][64 + d];
                float4 v1 = *(const float4*)&kvs[buf][jp + 1][64 + d];
#else
                const float* p0 = kvbase + (size_t)(j0 + jp) * kvrow;
                const float* p1 = kvbase + (size_t)(j0 + jp + 1) * kvrow;
                float4 k0 = *(const float4*)(p0 + d);
                float4 k1 = *(const float4*)(p1 + d);
                float4 v0 = *(const float4*)(p0 + 64 + d);
                float4 v1 = *(const float4*)(p1 + 64 + d);
#endif
                *(uint2*)&kl[jp][d]     = pk4(k0);
                *(uint2*)&kl[jp + 1][d] = pk4(k1);
                *(unsigned*)&vt[d + 0][jp] = pk2(v0.x, v1.x);
                *(unsigned*)&vt[d + 1][jp] = pk2(v0.y, v1.y);
                *(unsigned*)&vt[d + 2][jp] = pk2(v0.z, v1.z);
                *(unsigned*)&vt[d + 3][jp] = pk2(v0.w, v1.w);
            }
        }
#if HAVE_ASYNC
        if (jc < jcmax) {                          // prefetch next chunk
            int jn = (jc + 1) * 32, nb = buf ^ 1;
            #pragma unroll
            for (int i = 0; i < 8; ++i) {
                int idx = tid + i * 128;
                int key = idx >> 5, d4 = (idx & 31) * 4;
                async_b128(kvbase + (size_t)(jn + key) * kvrow + d4, &kvs[nb][key][d4]);
            }
        }
#endif
        __syncthreads();

        // S = q @ k^T (two 16-key tiles)
        v8f s0 = {}, s1 = {};
        s0 = wmma_bf16(aq0, ld16(&kl[lo][kb]),           s0);
        s0 = wmma_bf16(aq1, ld16(&kl[lo][32 + kb]),      s0);
        s1 = wmma_bf16(aq0, ld16(&kl[16 + lo][kb]),      s1);
        s1 = wmma_bf16(aq1, ld16(&kl[16 + lo][32 + kb]), s1);

        // online softmax
        float al[8];
        #pragma unroll
        for (int v = 0; v < 8; ++v) {
            int i  = i0 + v + 8 * lh;
            int ja = j0 + lo, jb = j0 + 16 + lo;
            float x0 = s0[v] * scale; if (ja > i + MLEN) x0 = -3.0e38f;
            float x1 = s1[v] * scale; if (jb > i + MLEN) x1 = -3.0e38f;
            float mx = fmaxf(x0, x1);
            #pragma unroll
            for (int msk = 1; msk < 16; msk <<= 1)
                mx = fmaxf(mx, __shfl_xor(mx, msk, 16));
            float mnew = fmaxf(rmax[v], mx);
            float p0 = __expf(x0 - mnew), p1 = __expf(x1 - mnew);
            float ps = p0 + p1;
            #pragma unroll
            for (int msk = 1; msk < 16; msk <<= 1)
                ps += __shfl_xor(ps, msk, 16);
            al[v]   = __expf(rmax[v] - mnew);
            rsum[v] = rsum[v] * al[v] + ps;
            rmax[v] = mnew;
            int row = v + 8 * lh;
            pl[wave][row][lo]      = tobf(p0);
            pl[wave][row][16 + lo] = tobf(p1);
        }
        #pragma unroll
        for (int v = 0; v < 8; ++v) {
            o0[v] *= al[v]; o1[v] *= al[v]; o2[v] *= al[v]; o3[v] *= al[v];
        }
        __syncthreads();

        // O += P[16x32] @ V[32x64]
        v16bf ap = ld16(&pl[wave][lo][kb]);
        o0 = wmma_bf16(ap, ld16(&vt[lo][kb]),      o0);
        o1 = wmma_bf16(ap, ld16(&vt[16 + lo][kb]), o1);
        o2 = wmma_bf16(ap, ld16(&vt[32 + lo][kb]), o2);
        o3 = wmma_bf16(ap, ld16(&vt[48 + lo][kb]), o3);
        buf ^= 1;
    }

    // normalize and write attn_vec
    #pragma unroll
    for (int v = 0; v < 8; ++v) {
        int i = i0 + v + 8 * lh;
        float inv = 1.0f / rsum[v];
        float* dst = obuf + ((size_t)i * BSZ + b) * (NHEAD * DHEAD) + hn * DHEAD;
        dst[lo]      = o0[v] * inv;
        dst[16 + lo] = o1[v] * inv;
        dst[32 + lo] = o2[v] * inv;
        dst[48 + lo] = o3[v] * inv;
    }
}

// ---------------------------------------------------------------------------
// h[row] = LayerNorm(h[row] + delta[row]) * g + b   (in place on h)
// ---------------------------------------------------------------------------
__global__ __launch_bounds__(256) void ln_residual_kernel(float* __restrict__ h,
                                                          const float* __restrict__ delta,
                                                          const float* __restrict__ g,
                                                          const float* __restrict__ bb) {
    __shared__ float r1[8], r2[8];
    const int row = blockIdx.x, tid = threadIdx.x;
    const int wave = tid >> 5, lane = tid & 31;
    float* hp = h + (size_t)row * DMODEL;
    const float* dp = delta + (size_t)row * DMODEL;

    float x[4];
    float s1 = 0.0f, s2 = 0.0f;
    #pragma unroll
    for (int e = 0; e < 4; ++e) {
        int d = tid * 4 + e;
        x[e] = hp[d] + dp[d];
        s1 += x[e];
        s2 += x[e] * x[e];
    }
    #pragma unroll
    for (int msk = 1; msk < 32; msk <<= 1) {
        s1 += __shfl_xor(s1, msk, 32);
        s2 += __shfl_xor(s2, msk, 32);
    }
    if (lane == 0) { r1[wave] = s1; r2[wave] = s2; }
    __syncthreads();
    float t1 = 0.0f, t2 = 0.0f;
    #pragma unroll
    for (int w = 0; w < 8; ++w) { t1 += r1[w]; t2 += r2[w]; }
    float mean = t1 * (1.0f / DMODEL);
    float var  = t2 * (1.0f / DMODEL) - mean * mean;
    float rstd = rsqrtf(var + LN_EPS);
    #pragma unroll
    for (int e = 0; e < 4; ++e) {
        int d = tid * 4 + e;
        hp[d] = (x[e] - mean) * rstd * g[d] + bb[d];
    }
}

// ---------------------------------------------------------------------------
// Host launcher
// ---------------------------------------------------------------------------
extern "C" void kernel_launch(void* const* d_in, const int* in_sizes, int n_in,
                              void* d_out, int out_size, void* d_ws, size_t ws_size,
                              hipStream_t stream) {
    (void)in_sizes; (void)n_in; (void)out_size; (void)ws_size;
    const int*   data = (const int*)  d_in[0];
    const float* mems = (const float*)d_in[1];
    const float* emb  = (const float*)d_in[2];
    const float* Wq   = (const float*)d_in[3];
    const float* Wkv  = (const float*)d_in[4];
    const float* Wo   = (const float*)d_in[5];
    const float* ln1g = (const float*)d_in[6];
    const float* ln1b = (const float*)d_in[7];
    const float* W1   = (const float*)d_in[8];
    const float* b1   = (const float*)d_in[9];
    const float* W2   = (const float*)d_in[10];
    const float* b2   = (const float*)d_in[11];
    const float* ln2g = (const float*)d_in[12];
    const float* ln2b = (const float*)d_in[13];

    const size_t NTOK  = (size_t)QLEN * BSZ;           // 2048 rows
    const size_t NCTX  = (size_t)KLEN * BSZ;           // 4096 rows
    const size_t SZ_H  = NTOK * DMODEL;                // 2M floats
    const size_t SZ_C  = NCTX * DMODEL;                // 4M floats
    const size_t SZ_KV = NCTX * 2 * DMODEL;            // 8M floats

    float* h    = (float*)d_ws;
    float* c    = h + SZ_H;          // reused as attention output
    float* q    = c + SZ_C;
    float* kv   = q + SZ_H;          // reused as FFN hidden
    float* tmp  = kv + SZ_KV;
    float* attn = c;
    float* ff   = kv;

    embed_kernel<<<dim3(QLEN, BSZ), 256, 0, stream>>>(data, emb, h);

    for (int l = 0; l < NLAYER; ++l) {
        (void)hipMemcpyAsync(c, mems + (size_t)l * (MLEN * BSZ * DMODEL),
                             (size_t)MLEN * BSZ * DMODEL * sizeof(float),
                             hipMemcpyDeviceToDevice, stream);
        (void)hipMemcpyAsync(c + (size_t)MLEN * BSZ * DMODEL, h,
                             SZ_H * sizeof(float), hipMemcpyDeviceToDevice, stream);

        // q = h @ Wq[l]            (2048 x 1024 x 1024)
        gemm_bf16<false, false><<<dim3(DMODEL / 64, NTOK / 128), 256, 0, stream>>>(
            h, Wq + (size_t)l * DMODEL * DMODEL, nullptr, q, (int)NTOK, DMODEL, DMODEL);

        // kv = c @ Wkv[l]          (4096 x 2048 x 1024)
        gemm_bf16<false, false><<<dim3(2 * DMODEL / 64, NCTX / 128), 256, 0, stream>>>(
            c, Wkv + (size_t)l * DMODEL * 2 * DMODEL, nullptr, kv, (int)NCTX, 2 * DMODEL, DMODEL);

        // fused masked softmax attention (4 q-tiles per block share KV chunks)
        attn_kernel<<<dim3(QLEN / 64, BSZ, NHEAD), 128, 0, stream>>>(q, kv, attn);

        // tmp = attn @ Wo[l]
        gemm_bf16<false, false><<<dim3(DMODEL / 64, NTOK / 128), 256, 0, stream>>>(
            attn, Wo + (size_t)l * DMODEL * DMODEL, nullptr, tmp, (int)NTOK, DMODEL, DMODEL);

        ln_residual_kernel<<<(int)NTOK, 256, 0, stream>>>(h, tmp, ln1g + l * DMODEL, ln1b + l * DMODEL);

        // ff = relu(h @ W1[l] + b1[l])   (2048 x 4096 x 1024)
        gemm_bf16<true, true><<<dim3(DINNER / 64, NTOK / 128), 256, 0, stream>>>(
            h, W1 + (size_t)l * DMODEL * DINNER, b1 + (size_t)l * DINNER, ff, (int)NTOK, DINNER, DMODEL);

        // tmp = ff @ W2[l] + b2[l]       (2048 x 1024 x 4096)
        gemm_bf16<true, false><<<dim3(DMODEL / 64, NTOK / 128), 256, 0, stream>>>(
            ff, W2 + (size_t)l * DINNER * DMODEL, b2 + (size_t)l * DMODEL, tmp, (int)NTOK, DMODEL, DINNER);

        ln_residual_kernel<<<(int)NTOK, 256, 0, stream>>>(h, tmp, ln2g + l * DMODEL, ln2b + l * DMODEL);
    }

    (void)hipMemcpyAsync(d_out, h, SZ_H * sizeof(float), hipMemcpyDeviceToDevice, stream);
}
